// TransformerBlock_69535520522492
// MI455X (gfx1250) — compile-verified, matrix operand order checked by reading, not run on hardware
//
#include <hip/hip_runtime.h>
#include <math.h>

// ---------------- WMMA types (CDNA5 / gfx1250, wave32) ----------------
typedef __attribute__((ext_vector_type(16))) __bf16 v16bf;
typedef __attribute__((ext_vector_type(8)))  float  v8f;

#define DIM 128
#define HID 512

// padded LDS strides (in bf16 elements) to avoid 64-bank conflicts
#define W1S 136   // W1 tile rows: [512][128] -> stride 136
#define W2S 520   // W2 tile rows: [128][512] -> stride 520
#define SHS 136   // h2 tile rows: [128][128] -> stride 136

// workspace layout (float index units)
#define CAP       16384
#define Q0_OFF    0
#define SKIP0_OFF 128
#define PROJ_OFF  256
#define CNT_OFF   384
#define EID_OFF   512
#define LOG_OFF   (512 + CAP)

static constexpr size_t SMEM_BYTES =
    (size_t)(512 * W1S + 128 * W2S + 128 * SHS + 8 * 16 * 32) * 2  // bf16 arrays
    + (size_t)(512 + 128 + 128 + 128 + 128) * 4;                    // f32 arrays

// Build a 16-element bf16 fragment following the CDNA5 16-bit A/B VGPR layout:
// elements 0..7  -> k = base + hi8 + e       (K 0..7 for lanes 0-15, 8..15 for 16-31)
// elements 8..15 -> k = base + 16 + hi8 + e  (K 16..23 / 24..31)
__device__ __forceinline__ v16bf load_frag(const __bf16* rp, int hi8) {
  v16bf r;
#pragma unroll
  for (int e = 0; e < 8; ++e) r[e] = rp[hi8 + e];
#pragma unroll
  for (int e = 0; e < 8; ++e) r[8 + e] = rp[16 + hi8 + e];
  return r;
}

__device__ __forceinline__ float gelu_exact(float v) {
  return 0.5f * v * (1.0f + erff(v * 0.70710678118654752f));
}

// ---------------- Stage A0: zero compaction counter ----------------
__global__ void k_init(float* ws) {
  if (threadIdx.x == 0) ((int*)(ws + CNT_OFF))[0] = 0;
}

// ---------------- Stage A1: LN(x[0]) -> q0, skip0 ----------------
__global__ void k_node0(const float* __restrict__ x,
                        const float* __restrict__ ln1_g, const float* __restrict__ ln1_b,
                        const float* __restrict__ Wq, const float* __restrict__ bq,
                        const float* __restrict__ Wskip, const float* __restrict__ bskip,
                        float* __restrict__ ws) {
  __shared__ float red[DIM], h0[DIM];
  int d = threadIdx.x;  // 128 threads
  float v = x[d];
  red[d] = v; __syncthreads();
  for (int s = 64; s > 0; s >>= 1) { if (d < s) red[d] += red[d + s]; __syncthreads(); }
  float mean = red[0] * (1.0f / DIM); __syncthreads();
  float c = v - mean;
  red[d] = c * c; __syncthreads();
  for (int s = 64; s > 0; s >>= 1) { if (d < s) red[d] += red[d + s]; __syncthreads(); }
  float rstd = rsqrtf(red[0] * (1.0f / DIM) + 1e-5f); __syncthreads();
  h0[d] = c * rstd * ln1_g[d] + ln1_b[d];
  __syncthreads();
  float q = bq[d], sk = bskip[d];
  for (int k = 0; k < DIM; ++k) {
    float h = h0[k];
    q  += Wq[d * DIM + k] * h;
    sk += Wskip[d * DIM + k] * h;
  }
  ws[Q0_OFF + d] = q;
  ws[SKIP0_OFF + d] = sk;
}

// ---------------- Stage A2: compact edges with dst == 0 ----------------
__global__ void k_compact(const long long* __restrict__ ei, int E, float* __restrict__ ws) {
  int e = blockIdx.x * 256 + threadIdx.x;
  if (e < E && ei[(long long)E + e] == 0) {
    int* cnt = (int*)(ws + CNT_OFF);
    int p = atomicAdd(cnt, 1);
    if (p < CAP) ((int*)(ws + EID_OFF))[p] = e;
  }
}

// ---------------- Stage A3: softmax attention for node 0 -> proj ----------------
__global__ void k_attn0(const float* __restrict__ x, const float* __restrict__ edge_attr,
                        const long long* __restrict__ ei, int E,
                        const float* __restrict__ ln1_g, const float* __restrict__ ln1_b,
                        const float* __restrict__ Wk, const float* __restrict__ bk,
                        const float* __restrict__ Wv, const float* __restrict__ bv,
                        const float* __restrict__ We,
                        const float* __restrict__ Wproj, const float* __restrict__ bproj,
                        float* __restrict__ ws) {
  __shared__ float red[DIM], hs[DIM], ea[DIM], sout[DIM];
  __shared__ float smax[2], sden[2];
  int d = threadIdx.x;  // 128 threads
  const int* cnt  = (const int*)(ws + CNT_OFF);
  const int* eids = (const int*)(ws + EID_OFF);
  int M = *cnt; if (M > CAP) M = CAP;

  float q0   = ws[Q0_OFF + d];
  float out0 = ws[SKIP0_OFF + d];
  if (d < 2) smax[d] = -3.0e38f;
  __syncthreads();

  // pass 1: logits + running max
  for (int i = 0; i < M; ++i) {
    int e = eids[i];
    long long s = ei[e];
    float xv = x[s * DIM + d];
    red[d] = xv; __syncthreads();
    for (int t = 64; t > 0; t >>= 1) { if (d < t) red[d] += red[d + t]; __syncthreads(); }
    float mean = red[0] * (1.0f / DIM); __syncthreads();
    float c = xv - mean;
    red[d] = c * c; __syncthreads();
    for (int t = 64; t > 0; t >>= 1) { if (d < t) red[d] += red[d + t]; __syncthreads(); }
    float rstd = rsqrtf(red[0] * (1.0f / DIM) + 1e-5f); __syncthreads();
    hs[d] = c * rstd * ln1_g[d] + ln1_b[d];
    ea[d] = edge_attr[(long long)e * DIM + d];
    __syncthreads();
    float kd = bk[d], ed = 0.0f;
    for (int k = 0; k < DIM; ++k) {
      kd += Wk[d * DIM + k] * hs[k];
      ed += We[d * DIM + k] * ea[k];
    }
    red[d] = q0 * (kd + ed); __syncthreads();
    for (int t = 32; t > 0; t >>= 1) { if ((d & 63) < t) red[d] += red[d + t]; __syncthreads(); }
    if (d == 0)  { float l = red[0]  * 0.125f; ws[LOG_OFF + 2 * i + 0] = l; if (l > smax[0]) smax[0] = l; }
    if (d == 64) { float l = red[64] * 0.125f; ws[LOG_OFF + 2 * i + 1] = l; if (l > smax[1]) smax[1] = l; }
    __syncthreads();
  }

  if (d < 2) {
    float den = 0.0f, mx = smax[d];
    for (int i = 0; i < M; ++i) den += expf(ws[LOG_OFF + 2 * i + d] - mx);
    sden[d] = den;
  }
  __syncthreads();

  // pass 2: accumulate out0 = sum alpha * (v[src] + e)  (+ skip already in out0)
  for (int i = 0; i < M; ++i) {
    int e = eids[i];
    long long s = ei[e];
    float xv = x[s * DIM + d];
    red[d] = xv; __syncthreads();
    for (int t = 64; t > 0; t >>= 1) { if (d < t) red[d] += red[d + t]; __syncthreads(); }
    float mean = red[0] * (1.0f / DIM); __syncthreads();
    float c = xv - mean;
    red[d] = c * c; __syncthreads();
    for (int t = 64; t > 0; t >>= 1) { if (d < t) red[d] += red[d + t]; __syncthreads(); }
    float rstd = rsqrtf(red[0] * (1.0f / DIM) + 1e-5f); __syncthreads();
    hs[d] = c * rstd * ln1_g[d] + ln1_b[d];
    ea[d] = edge_attr[(long long)e * DIM + d];
    __syncthreads();
    float vd = bv[d], ed = 0.0f;
    for (int k = 0; k < DIM; ++k) {
      vd += Wv[d * DIM + k] * hs[k];
      ed += We[d * DIM + k] * ea[k];
    }
    int h = d >> 6;
    float alpha = expf(ws[LOG_OFF + 2 * i + h] - smax[h]) / sden[h];
    out0 += (vd + ed) * alpha;
    __syncthreads();
  }

  sout[d] = out0; __syncthreads();
  float p = bproj[d];
  for (int k = 0; k < DIM; ++k) p += Wproj[d * DIM + k] * sout[k];
  ws[PROJ_OFF + d] = p;
}

// ---------------- Stage B: fused  x+proj -> LN -> MLP(WMMA bf16) -> +residual ----------------
__global__ __launch_bounds__(256) void k_mlp(const float* __restrict__ x,
                                             const float* __restrict__ ln2_g,
                                             const float* __restrict__ ln2_b,
                                             const float* __restrict__ W1, const float* __restrict__ b1,
                                             const float* __restrict__ W2, const float* __restrict__ b2,
                                             const float* __restrict__ ws,
                                             float* __restrict__ out, int N) {
  extern __shared__ char smem[];
  __bf16* sW1    = (__bf16*)smem;              // [512][W1S]
  __bf16* sW2    = sW1 + 512 * W1S;            // [128][W2S]
  __bf16* sH     = sW2 + 128 * W2S;            // [128][SHS]  h2 tile (bf16)
  __bf16* sStage = sH + 128 * SHS;             // 8 waves x [16][32]
  float*  sB1    = (float*)(sStage + 8 * 16 * 32);
  float*  sB2    = sB1 + HID;
  float*  sProj  = sB2 + DIM;
  float*  sG     = sProj + DIM;
  float*  sB     = sG + DIM;

  int tid = threadIdx.x;

  // -------- load weights / vectors into LDS (bf16 weights) --------
  for (int i = tid; i < HID * DIM; i += 256) {
    int o = i >> 7, k = i & 127;
    sW1[o * W1S + k] = (__bf16)W1[i];
  }
  for (int i = tid; i < DIM * HID; i += 256) {
    int dd = i >> 9, h = i & 511;
    sW2[dd * W2S + h] = (__bf16)W2[i];
  }
  for (int i = tid; i < HID; i += 256) sB1[i] = b1[i];
  if (tid < DIM) {
    sB2[tid]   = b2[tid];
    sProj[tid] = ws[PROJ_OFF + tid];
    sG[tid]    = ln2_g[tid];
    sB[tid]    = ln2_b[tid];
  }
  __syncthreads();

  int nodeBase = blockIdx.x * 128;

  // -------- LN phase: 2 threads per row, 64 cols each --------
  {
    int r = tid >> 1, half = tid & 1;
    int node = nodeBase + r;
    const float* xr = x + (long long)node * DIM + half * 64;
    float sum = 0.0f, sq = 0.0f;
    if (node < N) {
#pragma unroll
      for (int j = 0; j < 16; ++j) {
        float4 v = ((const float4*)xr)[j];
        int cb = half * 64 + j * 4;
        float a0 = v.x + sProj[cb + 0], a1 = v.y + sProj[cb + 1];
        float a2 = v.z + sProj[cb + 2], a3 = v.w + sProj[cb + 3];
        sum += a0 + a1 + a2 + a3;
        sq  += a0 * a0 + a1 * a1 + a2 * a2 + a3 * a3;
      }
    }
    sum += __shfl_xor(sum, 1, 32);
    sq  += __shfl_xor(sq, 1, 32);
    float mean = sum * (1.0f / DIM);
    float var  = sq * (1.0f / DIM) - mean * mean;
    float rstd = rsqrtf(var + 1e-5f);
    __bf16* hr = sH + r * SHS + half * 64;
    if (node < N) {
#pragma unroll
      for (int j = 0; j < 16; ++j) {
        float4 v = ((const float4*)xr)[j];
        int cb = half * 64 + j * 4;
        hr[j * 4 + 0] = (__bf16)(((v.x + sProj[cb + 0]) - mean) * rstd * sG[cb + 0] + sB[cb + 0]);
        hr[j * 4 + 1] = (__bf16)(((v.y + sProj[cb + 1]) - mean) * rstd * sG[cb + 1] + sB[cb + 1]);
        hr[j * 4 + 2] = (__bf16)(((v.z + sProj[cb + 2]) - mean) * rstd * sG[cb + 2] + sB[cb + 2]);
        hr[j * 4 + 3] = (__bf16)(((v.w + sProj[cb + 3]) - mean) * rstd * sG[cb + 3] + sB[cb + 3]);
      }
    } else {
#pragma unroll
      for (int j = 0; j < 64; ++j) hr[j] = (__bf16)0.0f;
    }
  }
  __syncthreads();

  // -------- WMMA phase: wave -> 16-row tile; full 512-hidden in 16 chunks --------
  int lane = tid & 31;
  int wave = tid >> 5;
  int rowBase = wave * 16;
  int hi8  = (lane >> 4) << 3;  // 0 for lanes 0-15, 8 for 16-31
  int ln15 = lane & 15;
  __bf16* stg = sStage + wave * (16 * 32);

  // A fragments of h2 (16 rows x K=128)
  v16bf aH[4];
#pragma unroll
  for (int kc = 0; kc < 4; ++kc)
    aH[kc] = load_frag(sH + (rowBase + ln15) * SHS + kc * 32, hi8);

  // GEMM2 accumulators: 8 col tiles x 16x16, init with residual x+proj and b2
  v8f acc[8];
#pragma unroll
  for (int t = 0; t < 8; ++t) {
    int col = t * 16 + ln15;
    float base = sB2[col] + sProj[col];
#pragma unroll
    for (int g = 0; g < 8; ++g) {
      int node = nodeBase + rowBase + hi8 + g;
      float xv = (node < N) ? x[(long long)node * DIM + col] : 0.0f;
      acc[t][g] = xv + base;
    }
  }

#pragma unroll 1
  for (int c = 0; c < 16; ++c) {
    // ---- GEMM1: hid[:, c*32 .. c*32+31] = h2 @ W1.T (+b1), then GELU ----
#pragma unroll
    for (int j = 0; j < 2; ++j) {
      int o0 = c * 32 + j * 16;
      float bval = sB1[o0 + ln15];
      v8f cc = {bval, bval, bval, bval, bval, bval, bval, bval};
#pragma unroll
      for (int kc = 0; kc < 4; ++kc) {
        v16bf bf = load_frag(sW1 + (o0 + ln15) * W1S + kc * 32, hi8);
        cc = __builtin_amdgcn_wmma_f32_16x16x32_bf16(false, aH[kc], false, bf,
                                                     (short)0, cc, false, false);
      }
#pragma unroll
      for (int g = 0; g < 8; ++g)
        stg[(hi8 + g) * 32 + j * 16 + ln15] = (__bf16)gelu_exact(cc[g]);
    }
    // ---- D-layout -> A-layout via wave-private LDS staging ----
    v16bf a2 = load_frag(stg + ln15 * 32, hi8);
    // ---- GEMM2: acc += act @ W2.T ----
#pragma unroll
    for (int t = 0; t < 8; ++t) {
      v16bf bf = load_frag(sW2 + (t * 16 + ln15) * W2S + c * 32, hi8);
      acc[t] = __builtin_amdgcn_wmma_f32_16x16x32_bf16(false, a2, false, bf,
                                                       (short)0, acc[t], false, false);
    }
  }

  // -------- store: out = x + proj + mlp --------
#pragma unroll
  for (int t = 0; t < 8; ++t) {
    int col = t * 16 + ln15;
#pragma unroll
    for (int g = 0; g < 8; ++g) {
      int node = nodeBase + rowBase + hi8 + g;
      if (node < N) out[(long long)node * DIM + col] = acc[t][g];
    }
  }
}

// ---------------- launcher ----------------
extern "C" void kernel_launch(void* const* d_in, const int* in_sizes, int n_in,
                              void* d_out, int out_size, void* d_ws, size_t ws_size,
                              hipStream_t stream) {
  const float*     x         = (const float*)d_in[0];
  const float*     edge_attr = (const float*)d_in[1];
  const long long* ei        = (const long long*)d_in[2];
  const float*     ln1_g = (const float*)d_in[3];
  const float*     ln1_b = (const float*)d_in[4];
  const float*     ln2_g = (const float*)d_in[5];
  const float*     ln2_b = (const float*)d_in[6];
  const float*     Wq = (const float*)d_in[7],  *bq = (const float*)d_in[8];
  const float*     Wk = (const float*)d_in[9],  *bk = (const float*)d_in[10];
  const float*     Wv = (const float*)d_in[11], *bv = (const float*)d_in[12];
  const float*     We = (const float*)d_in[13];
  const float*     Wskip = (const float*)d_in[14], *bskip = (const float*)d_in[15];
  const float*     Wproj = (const float*)d_in[16], *bproj = (const float*)d_in[17];
  const float*     W1 = (const float*)d_in[18], *b1 = (const float*)d_in[19];
  const float*     W2 = (const float*)d_in[20], *b2 = (const float*)d_in[21];
  float* out = (float*)d_out;
  float* ws  = (float*)d_ws;

  int N = in_sizes[0] / DIM;
  int E = in_sizes[1] / DIM;

  k_init<<<1, 32, 0, stream>>>(ws);
  k_node0<<<1, 128, 0, stream>>>(x, ln1_g, ln1_b, Wq, bq, Wskip, bskip, ws);
  k_compact<<<(E + 255) / 256, 256, 0, stream>>>(ei, E, ws);
  k_attn0<<<1, 128, 0, stream>>>(x, edge_attr, ei, E, ln1_g, ln1_b,
                                 Wk, bk, Wv, bv, We, Wproj, bproj, ws);
  int blocks = (N + 127) / 128;
  k_mlp<<<blocks, 256, SMEM_BYTES, stream>>>(x, ln2_g, ln2_b, W1, b1, W2, b2, ws, out, N);
}
